// MonitoredOptimizedAttention_28080496181582
// MI455X (gfx1250) — compile-verified, matrix operand order checked by reading, not run on hardware
//
#include <hip/hip_runtime.h>
#include <hip/hip_bf16.h>

// MHA forward for MI455X (gfx1250, wave32). All matmuls via
// v_wmma_f32_16x16x32_f16; attention is flash-style (online softmax),
// staging P through LDS to convert C-layout -> A-layout fragments.
// Projection GEMMs use 32x64 per-wave register blocking (8 C fragments)
// for ~21 FLOP/byte of L2 traffic instead of 8.
//
// Dims (fixed by the reference): B=2, T=2048, D=1024, H=16, Dh=64, M=B*T=4096.

typedef __attribute__((ext_vector_type(8)))  float    v8f;
typedef __attribute__((ext_vector_type(8)))  _Float16 v8h;
typedef __attribute__((ext_vector_type(16))) _Float16 v16h;

#define D_MODEL 1024
#define N_HEADS 16
#define HEAD_DIM 64
#define BATCH 2
#define SEQ 2048
#define MTOT (BATCH * SEQ)   // 4096

// ---------------------------------------------------------------------------
// Fragment loader for the CDNA5 16-bit A/B WMMA layout (16x32 / 32x16).
// Per ISA 7.12.2: lane<16 holds K = {0..7, 16..23}; lane>=16 holds
// K = {8..15, 24..31}. Caller passes (rowbase + k0) already offset so the
// fragment is two 16-byte contiguous chunks at +kb and +kb+16 halves.
// ---------------------------------------------------------------------------
__device__ __forceinline__ v16h load_frag(const _Float16* rowbase_k0, int lane) {
  const _Float16* p = rowbase_k0 + ((lane & 16) ? 8 : 0);
  v8h lo = *(const v8h*)(p);
  v8h hi = *(const v8h*)(p + 16);
  return __builtin_shufflevector(lo, hi, 0,1,2,3,4,5,6,7,8,9,10,11,12,13,14,15);
}

__device__ __forceinline__ v8f wmma_f16(v16h a, v16h b, v8f c) {
  return __builtin_amdgcn_wmma_f32_16x16x32_f16(
      /*neg_a=*/false, a, /*neg_b=*/false, b,
      /*c_mod=*/(short)0, c, /*reuse_a=*/false, /*reuse_b=*/false);
}

// ---------------------------------------------------------------------------
// f32 -> f16 elementwise convert
// ---------------------------------------------------------------------------
__global__ void cvt_f32_to_f16_kernel(const float* __restrict__ in,
                                      _Float16* __restrict__ out, int n) {
  int i = blockIdx.x * blockDim.x + threadIdx.x;
  int stride = gridDim.x * blockDim.x;
  for (; i < n; i += stride) out[i] = (_Float16)in[i];
}

// ---------------------------------------------------------------------------
// GEMM: Y = A(f16, M x 1024) @ W(f16, 1024 x 1024, row n holds W[n,:])^T + bias
// Each wave computes a 32x64 output tile: 2 (M) x 4 (N) fragments of 16x16.
// Per 32-wide k-step: 2 A-frags + 4 B-frags -> 8 WMMAs (A and B both reused).
// 4 waves per block cover a 32x256 block tile; grid = (M/32, N/256).
// mode 0: rotate_half epilogue, scatter f16 to (B,H,T,Dh)   (Q, K)
// mode 1: no rotate, scatter f16 to (B,H,Dh,T) transposed   (V)
// mode 2: no rotate, write f32 row-major M x 1024           (final out-proj)
// ---------------------------------------------------------------------------
__global__ void gemm_proj_kernel(const _Float16* __restrict__ A,
                                 const _Float16* __restrict__ W,
                                 const float* __restrict__ bias,
                                 void* __restrict__ outp, int mode) {
  const int lane = threadIdx.x & 31;
  const int wave = threadIdx.x >> 5;
  const int nl   = lane & 15;
  const int m0   = blockIdx.x * 32;
  const int n0   = (blockIdx.y * 4 + wave) * 64;

  const _Float16* arow0 = A + (size_t)(m0 + nl) * D_MODEL;
  const _Float16* arow1 = A + (size_t)(m0 + 16 + nl) * D_MODEL;
  const _Float16* wrow0 = W + (size_t)(n0 + nl) * D_MODEL;
  const _Float16* wrow1 = W + (size_t)(n0 + 16 + nl) * D_MODEL;
  const _Float16* wrow2 = W + (size_t)(n0 + 32 + nl) * D_MODEL;
  const _Float16* wrow3 = W + (size_t)(n0 + 48 + nl) * D_MODEL;

  v8f c[2][4];
#pragma unroll
  for (int mi = 0; mi < 2; ++mi)
#pragma unroll
    for (int fi = 0; fi < 4; ++fi) c[mi][fi] = (v8f){};

  for (int k0 = 0; k0 < D_MODEL; k0 += 32) {
    // hint next k-chunk of the A rows into cache (global_prefetch_b8)
    if (k0 + 256 < D_MODEL) {
      __builtin_prefetch(arow0 + k0 + 256, 0, 1);
      __builtin_prefetch(arow1 + k0 + 256, 0, 1);
    }
    v16h a0 = load_frag(arow0 + k0, lane);
    v16h a1 = load_frag(arow1 + k0, lane);
    v16h b0 = load_frag(wrow0 + k0, lane);
    v16h b1 = load_frag(wrow1 + k0, lane);
    v16h b2 = load_frag(wrow2 + k0, lane);
    v16h b3 = load_frag(wrow3 + k0, lane);
    c[0][0] = wmma_f16(a0, b0, c[0][0]);
    c[1][0] = wmma_f16(a1, b0, c[1][0]);
    c[0][1] = wmma_f16(a0, b1, c[0][1]);
    c[1][1] = wmma_f16(a1, b1, c[1][1]);
    c[0][2] = wmma_f16(a0, b2, c[0][2]);
    c[1][2] = wmma_f16(a1, b2, c[1][2]);
    c[0][3] = wmma_f16(a0, b3, c[0][3]);
    c[1][3] = wmma_f16(a1, b3, c[1][3]);
  }

#pragma unroll
  for (int mi = 0; mi < 2; ++mi) {
#pragma unroll
    for (int fi = 0; fi < 4; ++fi) {
      const int n  = n0 + fi * 16 + nl;
      const float bi = bias[n];
#pragma unroll
      for (int r = 0; r < 8; ++r) {
        const int m = m0 + mi * 16 + r + ((lane & 16) ? 8 : 0);
        float y = c[mi][fi][r] + bi;
        if (mode == 2) {
          ((float*)outp)[(size_t)m * D_MODEL + n] = y;
        } else {
          const int b  = m >> 11;         // m / SEQ
          const int t  = m & (SEQ - 1);
          const int hh = n >> 6;          // n / HEAD_DIM
          const int dh = n & (HEAD_DIM - 1);
          _Float16* dst = (_Float16*)outp;
          if (mode == 0) {
            // rotate_half: pairs (x0,x1) -> (-x1, x0); partner column = n^1
            // lives in lane^1 of the same 16-lane half of the C layout.
            float part = __shfl_xor(y, 1, 32);
            float val  = (n & 1) ? part : -part;
            dst[(((size_t)(b * N_HEADS + hh)) * SEQ + t) * HEAD_DIM + dh] = (_Float16)val;
          } else { // mode 1: V transposed (B,H,Dh,T)
            dst[(((size_t)(b * N_HEADS + hh)) * HEAD_DIM + dh) * SEQ + t] = (_Float16)y;
          }
        }
      }
    }
  }
}

// ---------------------------------------------------------------------------
// Flash attention: each wave owns a 16-row query block for one (b,h).
// Streams 32-key blocks: S = Q K^T via WMMA, online softmax in registers,
// P staged through LDS (C-layout -> A-layout), O += P V via WMMA against
// transposed V. Writes O as f16 to (B,T,D) for the final projection.
// ---------------------------------------------------------------------------
__global__ void flash_attn_kernel(const _Float16* __restrict__ Qh,
                                  const _Float16* __restrict__ Kh,
                                  const _Float16* __restrict__ Vt,
                                  _Float16* __restrict__ Oh) {
  const int lane = threadIdx.x & 31;
  const int wave = threadIdx.x >> 5;
  const int nl   = lane & 15;
  const int hi   = (lane & 16) ? 8 : 0;
  const int bh   = blockIdx.y;                       // b*16 + h
  const int q0   = (blockIdx.x * 4 + wave) * 16;

  const _Float16* Qb = Qh + (size_t)bh * SEQ * HEAD_DIM;
  const _Float16* Kb = Kh + (size_t)bh * SEQ * HEAD_DIM;
  const _Float16* Vb = Vt + (size_t)bh * HEAD_DIM * SEQ;

  __shared__ _Float16 Pst[4][16 * 32];
  _Float16* Pl = &Pst[wave][0];

  // Q fragments: A-matrix rows q0..q0+15, K-dim = head dim (64 = 2 x 32)
  const _Float16* qrow = Qb + (size_t)(q0 + nl) * HEAD_DIM;
  v16h aq0 = load_frag(qrow, lane);
  v16h aq1 = load_frag(qrow + 32, lane);

  v8f o[4];
  float m8[8], l8[8];
#pragma unroll
  for (int f = 0; f < 4; ++f) o[f] = (v8f){};
#pragma unroll
  for (int r = 0; r < 8; ++r) { m8[r] = -__builtin_inff(); l8[r] = 0.0f; }

  const float scale = 0.125f;  // 1/sqrt(64)
  const int kend = q0 + 16;    // causal: keys <= query row

  for (int k0 = 0; k0 < kend; k0 += 32) {
    // --- S = Q K^T for two 16-key halves (B-frag lane n = K row k0+n) ---
    v8f s0 = {}, s1 = {};
    {
      const _Float16* krow0 = Kb + (size_t)(k0 + nl) * HEAD_DIM;
      s0 = wmma_f16(aq0, load_frag(krow0, lane), s0);
      s0 = wmma_f16(aq1, load_frag(krow0 + 32, lane), s0);
      const _Float16* krow1 = Kb + (size_t)(k0 + 16 + nl) * HEAD_DIM;
      s1 = wmma_f16(aq0, load_frag(krow1, lane), s1);
      s1 = wmma_f16(aq1, load_frag(krow1 + 32, lane), s1);
    }

    // --- online softmax per row (rows striped: lane half + vgpr index) ---
#pragma unroll
    for (int r = 0; r < 8; ++r) {
      const int row  = q0 + r + hi;
      const int col0 = k0 + nl;
      const int col1 = col0 + 16;
      float v0 = s0[r] * scale;
      float v1 = s1[r] * scale;
      if (col0 > row) v0 = -__builtin_inff();
      if (col1 > row) v1 = -__builtin_inff();

      float bm = fmaxf(v0, v1);
#pragma unroll
      for (int msk = 1; msk < 16; msk <<= 1) bm = fmaxf(bm, __shfl_xor(bm, msk, 32));

      const float mnew  = fmaxf(m8[r], bm);
      const float alpha = __expf(m8[r] - mnew);
      const float p0 = __expf(v0 - mnew);
      const float p1 = __expf(v1 - mnew);
      float rs = p0 + p1;
#pragma unroll
      for (int msk = 1; msk < 16; msk <<= 1) rs += __shfl_xor(rs, msk, 32);

      l8[r] = l8[r] * alpha + rs;
      m8[r] = mnew;

      const int mrow = r + hi;
      Pl[mrow * 32 + nl]      = (_Float16)p0;
      Pl[mrow * 32 + 16 + nl] = (_Float16)p1;

      o[0][r] *= alpha; o[1][r] *= alpha; o[2][r] *= alpha; o[3][r] *= alpha;
    }

    // all lanes' P stores must land before cross-lane A-fragment reload
    asm volatile("s_wait_dscnt 0" ::: "memory");

    // --- O += P V : A-frag of P from LDS, B-frag from transposed V ---
    v16h ap = load_frag(Pl + nl * 32, lane);
#pragma unroll
    for (int f = 0; f < 4; ++f) {
      const _Float16* vrow = Vb + (size_t)(f * 16 + nl) * SEQ + k0;
      o[f] = wmma_f16(ap, load_frag(vrow, lane), o[f]);
    }
  }

  // --- normalize and write O to (B,T,D) f16 ---
  const int b = bh >> 4;
  const int h = bh & 15;
#pragma unroll
  for (int r = 0; r < 8; ++r) {
    const float inv = 1.0f / l8[r];
    const int t = q0 + r + hi;
#pragma unroll
    for (int f = 0; f < 4; ++f) {
      float val = o[f][r] * inv;
      Oh[((size_t)(b * SEQ + t)) * D_MODEL + h * HEAD_DIM + f * 16 + nl] = (_Float16)val;
    }
  }
}

// ---------------------------------------------------------------------------
// Launch: convert -> QKV proj (+rotate) -> flash attention -> out proj.
// Workspace layout (48 MiB total, all f16):
//   xh 8M | Wqh 2M | Wkh 2M | Wvh 2M | Woh 2M | Qh 8M | Kh 8M | Vt 8M | Oh 8M
// ---------------------------------------------------------------------------
extern "C" void kernel_launch(void* const* d_in, const int* in_sizes, int n_in,
                              void* d_out, int out_size, void* d_ws, size_t ws_size,
                              hipStream_t stream) {
  const float* x  = (const float*)d_in[0];
  const float* Wq = (const float*)d_in[1];
  const float* bq = (const float*)d_in[2];
  const float* Wk = (const float*)d_in[3];
  const float* bk = (const float*)d_in[4];
  const float* Wv = (const float*)d_in[5];
  const float* bv = (const float*)d_in[6];
  const float* Wo = (const float*)d_in[7];
  const float* bo = (const float*)d_in[8];

  char* ws = (char*)d_ws;
  _Float16* xh  = (_Float16*)(ws);
  _Float16* Wqh = (_Float16*)(ws + (8u << 20));
  _Float16* Wkh = (_Float16*)(ws + (10u << 20));
  _Float16* Wvh = (_Float16*)(ws + (12u << 20));
  _Float16* Woh = (_Float16*)(ws + (14u << 20));
  _Float16* Qh  = (_Float16*)(ws + (16u << 20));
  _Float16* Kh  = (_Float16*)(ws + (24u << 20));
  _Float16* Vt  = (_Float16*)(ws + (32u << 20));
  _Float16* Oh  = (_Float16*)(ws + (40u << 20));

  // 1) precision convert (f32 -> f16)
  cvt_f32_to_f16_kernel<<<1024, 256, 0, stream>>>(x, xh, MTOT * D_MODEL);
  cvt_f32_to_f16_kernel<<<512, 256, 0, stream>>>(Wq, Wqh, D_MODEL * D_MODEL);
  cvt_f32_to_f16_kernel<<<512, 256, 0, stream>>>(Wk, Wkh, D_MODEL * D_MODEL);
  cvt_f32_to_f16_kernel<<<512, 256, 0, stream>>>(Wv, Wvh, D_MODEL * D_MODEL);
  cvt_f32_to_f16_kernel<<<512, 256, 0, stream>>>(Wo, Woh, D_MODEL * D_MODEL);

  // 2) QKV projections (rotate for Q/K; V stored transposed)
  dim3 ggrid(MTOT / 32, D_MODEL / 256);
  gemm_proj_kernel<<<ggrid, 128, 0, stream>>>(xh, Wqh, bq, (void*)Qh, 0);
  gemm_proj_kernel<<<ggrid, 128, 0, stream>>>(xh, Wkh, bk, (void*)Kh, 0);
  gemm_proj_kernel<<<ggrid, 128, 0, stream>>>(xh, Wvh, bv, (void*)Vt, 1);

  // 3) causal flash attention
  dim3 agrid(SEQ / 64, BATCH * N_HEADS);
  flash_attn_kernel<<<agrid, 128, 0, stream>>>(Qh, Kh, Vt, Oh);

  // 4) output projection -> fp32 d_out
  gemm_proj_kernel<<<ggrid, 128, 0, stream>>>(Oh, Woh, bo, d_out, 2);
}
